// HypothesisDecoder_48284022341789
// MI455X (gfx1250) — matrix operand order repaired and checked by reading, not compiled, not run on hardware
//
#include <hip/hip_runtime.h>
#include <hip/hip_bf16.h>
#include <math.h>

// ---------------- problem constants (match reference) ----------------
#define BATCH 2
#define GD    96
#define RESV  0.04f
#define EPSBN 1e-5f
#define NHYP  7
#define GRIDN (BATCH * GD * GD * GD)

// ---------------- CDNA5 WMMA types ----------------
typedef __attribute__((ext_vector_type(16))) __bf16 v16bf;
typedef __attribute__((ext_vector_type(8)))  float  v8f;

union ABfrag { unsigned int u[8]; v16bf v; };

// native bf16 convert (gfx1250 has hw bf16 cvt; RNE semantics match reference intent)
__device__ __forceinline__ unsigned short f2bf(float f) {
  __bf16 h = (__bf16)f;
  return __builtin_bit_cast(unsigned short, h);
}
__device__ __forceinline__ float bf2f(unsigned short h) {
  return __uint_as_float(((unsigned)h) << 16);
}

// ---------------- CDNA5 async global->LDS staging (guarded) ----------------
#if __has_builtin(__builtin_amdgcn_global_load_async_to_lds_b128)
#define HAS_ASYNC_B128 1
#else
#define HAS_ASYNC_B128 0
#endif
#if __has_builtin(__builtin_amdgcn_s_wait_asynccnt)
#define WAIT_ASYNC() __builtin_amdgcn_s_wait_asynccnt(0)
#else
#define WAIT_ASYNC() asm volatile("s_wait_asynccnt 0x0" ::: "memory")
#endif

// builtin parameter type per hipcc diagnostic: int __attribute__((vector_size(16)))*
typedef int async_v4i __attribute__((vector_size(16)));
#if HAS_ASYNC_B128
#define ASYNC_CP16(gsrc, ldst) \
  __builtin_amdgcn_global_load_async_to_lds_b128((async_v4i*)(gsrc), (async_v4i*)(ldst), 0, 0)
#endif

// linear LDS staging helper (ndw must be a multiple of 4)
__device__ __forceinline__ void stage_lds(unsigned int* __restrict__ dst,
                                          const unsigned int* __restrict__ src,
                                          int ndw, int tid) {
#if HAS_ASYNC_B128
  for (int i = tid * 4; i < ndw; i += 256 * 4)
    ASYNC_CP16(src + i, dst + i);
  WAIT_ASYNC();
#else
  for (int i = tid; i < ndw; i += 256) dst[i] = src[i];
#endif
}

// ---------------- LDS layout for the fused conv kernel ----------------
// 32 points/block. Activations stored with guard rows: 9 LDS rows per point
// (l = -1..7), pad rows zeroed once => branch-free im2col A fragments.
// x tile:  288 x (64+8)  bf16  = 41472 B
// actA:    288 x (128+8) bf16  = 78336 B
// actB:    288 x (128+8) bf16  = 78336 B
// weights: 128 x (384+8) bf16  = 100352 B (max layer; layer1 uses 128x200)
// logits:  224 f32             = 896 B
#define TP     32           // points per block
#define MTILES 14           // TP*7/16
#define XS0    72
#define ACTS   136
#define WS1    200          // 3*64 + 8
#define WS2    392          // 3*128 + 8
#define X_OFF   0
#define AA_OFF  41472
#define AB_OFF  119808
#define W_OFF   198144
#define LG_OFF  298496
#define SMEM_BYTES 299392

// ---------------- kernel 1: init grid + per-batch min ----------------
__global__ __launch_bounds__(256) void hd_init_kernel(int* __restrict__ grid,
                                                      unsigned int* __restrict__ minbits) {
  int i = blockIdx.x * blockDim.x + threadIdx.x;
  if (i < GRIDN) grid[i] = 0x7fffffff;
  if (i < BATCH * 3) minbits[i] = 0x7f7fffffu;   // FLT_MAX bits (coords >= 0)
}

// ---------------- kernel 2: scatter voxel rows + segment-min ----------------
__global__ __launch_bounds__(256) void hd_scatter_kernel(
    const float* __restrict__ spts, const int* __restrict__ scoords,
    const int* __restrict__ sbatch, int* __restrict__ grid,
    int* __restrict__ minbits, int n) {
  int i = blockIdx.x * blockDim.x + threadIdx.x;
  if (i >= n) return;
  int b = sbatch[i];
#pragma unroll
  for (int d = 0; d < 3; ++d)   // non-negative floats: int compare == float compare
    atomicMin(&minbits[b * 3 + d], __float_as_int(spts[i * 3 + d]));
  int x = scoords[i * 3 + 0], y = scoords[i * 3 + 1], z = scoords[i * 3 + 2];
  int gi = ((b * GD + x) * GD + y) * GD + z;
  atomicMin(&grid[gi], i);      // leftmost duplicate (matches stable sort + searchsorted)
}

// ---------------- kernel 3: weight transform (O,K padded, bf16) + BN fold ----------------
__global__ __launch_bounds__(256) void hd_prep_kernel(
    const float* __restrict__ w1, const float* __restrict__ w2,
    const float* __restrict__ w3, const float* __restrict__ w4,
    const float* __restrict__ g1, const float* __restrict__ b1,
    const float* __restrict__ m1, const float* __restrict__ v1,
    const float* __restrict__ g2, const float* __restrict__ b2,
    const float* __restrict__ m2, const float* __restrict__ v2,
    const float* __restrict__ g3, const float* __restrict__ b3,
    const float* __restrict__ m3, const float* __restrict__ v3,
    unsigned short* __restrict__ wt1, unsigned short* __restrict__ wt2,
    unsigned short* __restrict__ wt3, float* __restrict__ wt4,
    float* __restrict__ sb) {
  int i = blockIdx.x * blockDim.x + threadIdx.x;
  const int n1 = 128 * 192, n2 = 128 * 384;
  if (i < n1) {                                   // layer1: K = kk*64 + c
    int o = i / 192, k = i % 192, kk = k / 64, c = k % 64;
    wt1[o * WS1 + k] = f2bf(w1[(o * 64 + c) * 3 + kk]);
    return;
  }
  i -= n1;
  if (i < n2) {                                   // layer2
    int o = i / 384, k = i % 384, kk = k / 128, c = k % 128;
    wt2[o * WS2 + k] = f2bf(w2[(o * 128 + c) * 3 + kk]);
    return;
  }
  i -= n2;
  if (i < n2) {                                   // layer3
    int o = i / 384, k = i % 384, kk = k / 128, c = k % 128;
    wt3[o * WS2 + k] = f2bf(w3[(o * 128 + c) * 3 + kk]);
    return;
  }
  i -= n2;
  if (i < 384) {                                  // layer4 (O=1), fp32
    int kk = i / 128, c = i % 128;
    wt4[i] = w4[c * 3 + kk];
    return;
  }
  i -= 384;
  if (i < 3 * 128) {                              // fold BN: y = x*s + (b - m*s)
    int li = i / 128, o = i % 128;
    const float* gs[3] = {g1, g2, g3};
    const float* bs[3] = {b1, b2, b3};
    const float* ms[3] = {m1, m2, m3};
    const float* vs[3] = {v1, v2, v3};
    float s = gs[li][o] / sqrtf(vs[li][o] + EPSBN);
    sb[li * 256 + o]       = s;
    sb[li * 256 + 128 + o] = bs[li][o] - ms[li][o] * s;
  }
}

// ---------------- kernel 4: trilinear interp + concat -> bf16 act0 (N*7, 64) ----------------
__global__ __launch_bounds__(256) void hd_interp_kernel(
    const float* __restrict__ dmh, const float* __restrict__ pfeat,
    const int* __restrict__ pbatch, const float* __restrict__ sfeats,
    const int* __restrict__ grid, const float* __restrict__ minpts,
    unsigned short* __restrict__ act0, int ntot) {
  int m = blockIdx.x * blockDim.x + threadIdx.x;
  if (m >= ntot) return;
  int p = m / NHYP;
  int b = pbatch[p];
  float fr[3];
  int   fl[3];
#pragma unroll
  for (int d = 0; d < 3; ++d) {
    float q = (dmh[(size_t)m * 3 + d] - minpts[b * 3 + d]) * (1.0f / RESV);
    float f = floorf(q);
    fl[d] = (int)f;
    fr[d] = q - f;
  }
  float acc[32];
#pragma unroll
  for (int c = 0; c < 32; ++c) acc[c] = 0.f;
#pragma unroll
  for (int corner = 0; corner < 8; ++corner) {
    int dx = (corner >> 2) & 1, dy = (corner >> 1) & 1, dz = corner & 1;
    int cx = fl[0] + dx, cy = fl[1] + dy, cz = fl[2] + dz;
    if ((unsigned)cx < (unsigned)GD && (unsigned)cy < (unsigned)GD &&
        (unsigned)cz < (unsigned)GD) {
      int idx = grid[((b * GD + cx) * GD + cy) * GD + cz];
      if (idx != 0x7fffffff) {
        float w = (dx ? fr[0] : 1.f - fr[0]) * (dy ? fr[1] : 1.f - fr[1]) *
                  (dz ? fr[2] : 1.f - fr[2]);
        const float4* f4 = (const float4*)(sfeats + (size_t)idx * 32);
#pragma unroll
        for (int v = 0; v < 8; ++v) {
          float4 t = f4[v];
          acc[v * 4 + 0] += w * t.x; acc[v * 4 + 1] += w * t.y;
          acc[v * 4 + 2] += w * t.z; acc[v * 4 + 3] += w * t.w;
        }
      }
    }
  }
  unsigned int* o32 = (unsigned int*)(act0 + (size_t)m * 64);
#pragma unroll
  for (int c = 0; c < 16; ++c)
    o32[c] = (unsigned)f2bf(acc[2 * c]) | ((unsigned)f2bf(acc[2 * c + 1]) << 16);
  const float* pf = pfeat + (size_t)m * 32;
#pragma unroll
  for (int c = 0; c < 16; ++c)
    o32[16 + c] = (unsigned)f2bf(pf[2 * c]) | ((unsigned)f2bf(pf[2 * c + 1]) << 16);
}

// ---------------- WMMA im2col conv layer (per 32-point tile, 8 waves = 128 out ch) ----------------
// Logical row r = p*7 + l (224 rows = 14 M-tiles of 16); LDS row = p*9 + l + 1
// with zeroed guard rows at p*9 and p*9+8. K index k = kk*CIN + c,
// A(r,k) = act_lds[p*9 + l + kk, c] -- always in-bounds, branch-free.
// B fragments are loaded from LDS ONCE per layer and held in registers across
// all 14 M-tiles (removes 13/14 of the B-side LDS traffic).
template <int CIN>
__device__ __forceinline__ void conv_layer(
    const unsigned short* __restrict__ w_l,            // LDS weights (128, 3*CIN+8)
    const unsigned short* __restrict__ src, int ss,    // LDS act (9 rows/pt), stride shorts
    unsigned short* __restrict__ dst, int ds,
    const float* __restrict__ sb,                      // s[0..127], bias[128..255]
    int lane, int otile) {
  constexpr int K  = 3 * CIN;
  constexpr int KT = K / 32;
  constexpr int WS = K + 8;
  const int m15   = lane & 15;
  const int khalf = lane >> 4;
  const int ocol  = otile * 16 + m15;
  const float sc  = sb[ocol];
  const float bi  = sb[128 + ocol];
  const unsigned int* wrow = (const unsigned int*)(w_l + ocol * WS);

  // hoist all B fragments into registers (col = lane%16, K half by lane/16)
  ABfrag bfr[KT];
#pragma unroll
  for (int kt = 0; kt < KT; ++kt) {
    const uint4* bp = (const uint4*)(wrow + (kt * 16 + 4 * khalf));
    uint4 q0 = bp[0];
    uint4 q1 = bp[2];
    bfr[kt].u[0] = q0.x; bfr[kt].u[1] = q0.y; bfr[kt].u[2] = q0.z; bfr[kt].u[3] = q0.w;
    bfr[kt].u[4] = q1.x; bfr[kt].u[5] = q1.y; bfr[kt].u[6] = q1.z; bfr[kt].u[7] = q1.w;
  }

#pragma unroll 1
  for (int mt = 0; mt < MTILES; ++mt) {
    int row = mt * 16 + m15;
    int p   = row / 7;
    int l   = row - p * 7;
    int ar0 = p * 9 + l;              // + kk gives A source row (guards absorb pad)
    v8f acc = {0.f, 0.f, 0.f, 0.f, 0.f, 0.f, 0.f, 0.f};
#pragma unroll
    for (int kt = 0; kt < KT; ++kt) {
      const int k0 = kt * 32;
      const int kk = k0 / CIN;        // constant after unroll
      const int cb = k0 % CIN;
      const unsigned int* srow = (const unsigned int*)(src + (ar0 + kk) * ss);
      // A: lane<16 -> K 0..7 / 16..23 ; lane>=16 -> K 8..15 / 24..31 (16B aligned)
      const uint4* ap = (const uint4*)(srow + ((cb >> 1) + 4 * khalf));
      uint4 qa0 = ap[0];
      uint4 qa1 = ap[2];
      ABfrag a;
      a.u[0] = qa0.x; a.u[1] = qa0.y; a.u[2] = qa0.z; a.u[3] = qa0.w;
      a.u[4] = qa1.x; a.u[5] = qa1.y; a.u[6] = qa1.z; a.u[7] = qa1.w;
      acc = __builtin_amdgcn_wmma_f32_16x16x32_bf16(
          false, a.v, false, bfr[kt].v, (short)0, acc, false, false);
    }
    // epilogue: BN fold + ReLU + bf16 store. D: M = i + 8*(lane/16), N = lane%16
#pragma unroll
    for (int i = 0; i < 8; ++i) {
      int r  = mt * 16 + i + 8 * khalf;
      int pp = r / 7;
      int lr = r - pp * 7;
      float v = fmaxf(acc[i] * sc + bi, 0.f);
      dst[(pp * 9 + lr + 1) * ds + ocol] = f2bf(v);
    }
  }
}

// ---------------- kernel 5: fused conv1-3 (WMMA) + conv4 + softmax ----------------
__global__ __launch_bounds__(256) void hd_fused_kernel(
    const unsigned short* __restrict__ act0,   // (N*7, 64) bf16
    const unsigned short* __restrict__ wt1,    // (128, 200) bf16
    const unsigned short* __restrict__ wt2,    // (128, 392) bf16
    const unsigned short* __restrict__ wt3,    // (128, 392) bf16
    const float* __restrict__ wt4,             // (384) fp32, k = kk*128 + c
    const float* __restrict__ bias4,           // (1)
    const float* __restrict__ sb,              // (3, 2, 128)
    float* __restrict__ out,                   // (N, 7)
    int npts) {
  extern __shared__ char smem[];
  unsigned short* x_l = (unsigned short*)(smem + X_OFF);
  unsigned short* aA  = (unsigned short*)(smem + AA_OFF);
  unsigned short* aB  = (unsigned short*)(smem + AB_OFF);
  unsigned short* w_l = (unsigned short*)(smem + W_OFF);
  float*          lg  = (float*)(smem + LG_OFF);

  const int tid  = threadIdx.x;
  const int lane = tid & 31;
  const int wave = tid >> 5;
  const int gp0  = blockIdx.x * TP;
  const int mtot = npts * NHYP;

  {  // zero guard rows (rows p*9 and p*9+8 of each buffer; stay zero throughout)
    unsigned int* xz = (unsigned int*)x_l;
    for (int i = tid; i < 2 * TP * 36; i += 256) {
      int r = i / 36, c = i - r * 36;
      xz[((r >> 1) * 9 + (r & 1) * 8) * 36 + c] = 0u;
    }
    unsigned int* az = (unsigned int*)aA;
    unsigned int* bz = (unsigned int*)aB;
    for (int i = tid; i < 2 * TP * 68; i += 256) {
      int r = i / 68, c = i - r * 68;
      int rr = ((r >> 1) * 9 + (r & 1) * 8) * 68 + c;
      az[rr] = 0u;
      bz[rr] = 0u;
    }
  }
  {  // stage input tile: 224 logical rows x 32 dwords -> guarded layout
    const unsigned int* src = (const unsigned int*)act0;
    unsigned int* dst = (unsigned int*)x_l;
    bool full = (gp0 * NHYP + TP * NHYP) <= mtot;
#if HAS_ASYNC_B128
    if (full) {
      for (int i = tid; i < TP * 7 * 8; i += 256) {  // 16B chunks
        int r = i >> 3, c4 = (i & 7) * 4;
        int p = r / 7, lr = r - p * 7;
        ASYNC_CP16(src + (size_t)(gp0 * NHYP + r) * 32 + c4,
                   dst + (p * 9 + lr + 1) * 36 + c4);
      }
      WAIT_ASYNC();
    } else
#endif
    {
      for (int i = tid; i < TP * 7 * 32; i += 256) {
        int r = i >> 5, c = i & 31;
        int p = r / 7, lr = r - p * 7;
        int grow = gp0 * NHYP + r;
        dst[(p * 9 + lr + 1) * 36 + c] = (grow < mtot) ? src[(size_t)grow * 32 + c] : 0u;
      }
    }
  }
  stage_lds((unsigned int*)w_l, (const unsigned int*)wt1, 128 * (WS1 / 2), tid);
  __syncthreads();
  conv_layer<64>(w_l, x_l, XS0, aA, ACTS, sb + 0 * 256, lane, wave);
  __syncthreads();
  stage_lds((unsigned int*)w_l, (const unsigned int*)wt2, 128 * (WS2 / 2), tid);
  __syncthreads();
  conv_layer<128>(w_l, aA, ACTS, aB, ACTS, sb + 1 * 256, lane, wave);
  __syncthreads();
  stage_lds((unsigned int*)w_l, (const unsigned int*)wt3, 128 * (WS2 / 2), tid);
  __syncthreads();
  conv_layer<128>(w_l, aB, ACTS, aA, ACTS, sb + 2 * 256, lane, wave);
  __syncthreads();

  // layer 4: 128->1 conv (k=3) per (p,l) row; guard rows make it branch-free
  if (tid < TP * 7) {
    int p = tid / NHYP, l = tid - p * NHYP;
    float acc = bias4[0];
#pragma unroll
    for (int kk = 0; kk < 3; ++kk) {
      const unsigned int* row32 = (const unsigned int*)(aA + (p * 9 + l + kk) * ACTS);
      const float* wr = wt4 + kk * 128;
      for (int c2 = 0; c2 < 64; ++c2) {
        unsigned int d = row32[c2];
        acc += wr[2 * c2]     * bf2f((unsigned short)(d & 0xffffu));
        acc += wr[2 * c2 + 1] * bf2f((unsigned short)(d >> 16));
      }
    }
    lg[tid] = acc;
  }
  __syncthreads();

  // softmax over 7 hypotheses per point
  if (tid < TP) {
    int gp = gp0 + tid;
    if (gp < npts) {
      float mx = -3.4e38f;
#pragma unroll
      for (int j = 0; j < NHYP; ++j) mx = fmaxf(mx, lg[tid * NHYP + j]);
      float e[NHYP], s = 0.f;
#pragma unroll
      for (int j = 0; j < NHYP; ++j) { e[j] = expf(lg[tid * NHYP + j] - mx); s += e[j]; }
      float inv = 1.f / s;
#pragma unroll
      for (int j = 0; j < NHYP; ++j) out[(size_t)gp * NHYP + j] = e[j] * inv;
    }
  }
}

// ---------------- host launch ----------------
extern "C" void kernel_launch(void* const* d_in, const int* in_sizes, int n_in,
                              void* d_out, int out_size, void* d_ws, size_t ws_size,
                              hipStream_t stream) {
  const float* sparse_pts    = (const float*)d_in[0];
  const int*   sparse_coords = (const int*)d_in[1];
  const int*   sparse_batch  = (const int*)d_in[2];
  const float* sparse_feats  = (const float*)d_in[3];
  const float* dmh           = (const float*)d_in[4];
  const float* pfeat         = (const float*)d_in[5];
  const int*   pbatch        = (const int*)d_in[6];
  const float* w1 = (const float*)d_in[7];
  const float* g1 = (const float*)d_in[8];
  const float* b1 = (const float*)d_in[9];
  const float* m1 = (const float*)d_in[10];
  const float* v1 = (const float*)d_in[11];
  const float* w2 = (const float*)d_in[12];
  const float* g2 = (const float*)d_in[13];
  const float* b2 = (const float*)d_in[14];
  const float* m2 = (const float*)d_in[15];
  const float* v2 = (const float*)d_in[16];
  const float* w3 = (const float*)d_in[17];
  const float* g3 = (const float*)d_in[18];
  const float* b3 = (const float*)d_in[19];
  const float* m3 = (const float*)d_in[20];
  const float* v3 = (const float*)d_in[21];
  const float* w4 = (const float*)d_in[22];
  const float* bias4 = (const float*)d_in[23];

  const int n = in_sizes[2];       // sparse points
  const int N = in_sizes[6];       // decoded points
  const int M = N * NHYP;

  // workspace layout
  auto aln = [](size_t x) { return (x + 255) & ~(size_t)255; };
  char* ws = (char*)d_ws;
  size_t o_min  = 0;
  size_t o_grid = aln(o_min + BATCH * 3 * 4);
  size_t o_wt1  = aln(o_grid + (size_t)GRIDN * 4);
  size_t o_wt2  = aln(o_wt1 + (size_t)128 * WS1 * 2);
  size_t o_wt3  = aln(o_wt2 + (size_t)128 * WS2 * 2);
  size_t o_wt4  = aln(o_wt3 + (size_t)128 * WS2 * 2);
  size_t o_sb   = aln(o_wt4 + 384 * 4);
  size_t o_act0 = aln(o_sb + 3 * 256 * 4);

  int*            minb = (int*)(ws + o_min);
  int*            grid = (int*)(ws + o_grid);
  unsigned short* wt1  = (unsigned short*)(ws + o_wt1);
  unsigned short* wt2  = (unsigned short*)(ws + o_wt2);
  unsigned short* wt3  = (unsigned short*)(ws + o_wt3);
  float*          wt4  = (float*)(ws + o_wt4);
  float*          sbp  = (float*)(ws + o_sb);
  unsigned short* act0 = (unsigned short*)(ws + o_act0);
  float*          out  = (float*)d_out;

  (void)hipFuncSetAttribute((const void*)hd_fused_kernel,
                            hipFuncAttributeMaxDynamicSharedMemorySize, SMEM_BYTES);

  hd_init_kernel<<<(GRIDN + 255) / 256, 256, 0, stream>>>(grid, (unsigned int*)minb);
  hd_scatter_kernel<<<(n + 255) / 256, 256, 0, stream>>>(
      sparse_pts, sparse_coords, sparse_batch, grid, minb, n);
  const int prepN = 128 * 192 + 2 * 128 * 384 + 384 + 3 * 128;
  hd_prep_kernel<<<(prepN + 255) / 256, 256, 0, stream>>>(
      w1, w2, w3, w4, g1, b1, m1, v1, g2, b2, m2, v2, g3, b3, m3, v3,
      wt1, wt2, wt3, wt4, sbp);
  hd_interp_kernel<<<(M + 255) / 256, 256, 0, stream>>>(
      dmh, pfeat, pbatch, sparse_feats, grid, (const float*)minb, act0, M);
  hd_fused_kernel<<<(N + TP - 1) / TP, 256, SMEM_BYTES, stream>>>(
      act0, wt1, wt2, wt3, wt4, bias4, sbp, out, N);
}
